// RadarNet_51805895524445
// MI455X (gfx1250) — matrix operand order repaired and checked by reading, not compiled, not run on hardware
//
#include <hip/hip_runtime.h>
#include <hip/hip_bf16.h>

// ---------------------------------------------------------------------------
// Types for CDNA5 WMMA
// ---------------------------------------------------------------------------
typedef __attribute__((ext_vector_type(16))) __bf16 v16bf;
typedef __attribute__((ext_vector_type(8)))  float  v8f;

#define TOPK 10

__device__ __forceinline__ unsigned short f32_to_bf16_rne(float f) {
  union { float f; unsigned u; } x; x.f = f;
  unsigned r = x.u + 0x7FFFu + ((x.u >> 16) & 1u);
  return (unsigned short)(r >> 16);
}

// ---------------------------------------------------------------------------
// Weight packing: OIHW f32 -> bf16 B-fragments.
// K order matches the activation im2col rows: k = (ry*3+rx)*inC + ic.
// B-fragment lane layout (ISA 7.12.2): lanes 0-15 hold K=kc*32+{0..7,16..23} (N=lane),
// lanes 16-31 hold K=kc*32+{8..15,24..31}. storage: ((ct*nKC + kc)*32 + lane)*16 + e
// ---------------------------------------------------------------------------
__global__ void rnet_pack_w(const float* __restrict__ w, unsigned short* __restrict__ out,
                            int inC, int nKC, int outC) {
  int i = blockIdx.x * blockDim.x + threadIdx.x;
  int total = (outC >> 4) * nKC * 512;
  if (i >= total) return;
  int K = inC * 9;
  int e    = i & 15;
  int lane = (i >> 4) & 31;
  int kc   = (i >> 9) % nKC;
  int ct   = (i >> 9) / nKC;
  int k    = kc * 32 + ((lane >> 4) << 3) + (e & 7) + ((e >> 3) << 4);
  int n    = ct * 16 + (lane & 15);
  float val = 0.0f;
  if (k < K) {
    int rp = k / inC;        // (ry*3+rx)
    int ic = k % inC;
    val = w[(size_t)n * K + ic * 9 + rp];  // OIHW flat
  }
  out[i] = f32_to_bf16_rne(val);
}

__global__ void rnet_zero_f32(float* p, int n) {
  int i = blockIdx.x * blockDim.x + threadIdx.x;
  if (i < n) p[i] = 0.0f;
}

// NCHW -> NHWC for the network input (C=2)
__global__ void rnet_nchw_to_nhwc(const float* __restrict__ in, float* __restrict__ out,
                                  int C, int HW) {
  int i = blockIdx.x * blockDim.x + threadIdx.x;
  int total = 4 * C * HW;
  if (i >= total) return;
  int c = i % C;
  int j = i / C;            // b*HW + pix
  int pix = j % HW;
  int b = j / HW;
  out[i] = in[((size_t)(b * C + c)) * HW + pix];
}

// ---------------------------------------------------------------------------
// Fused Conv3x3 (implicit GEMM, v_wmma_f32_16x16x32_bf16) + BN partial stats.
// Activations NHWC. Block: 128 threads = 4 waves; one (b, y, 16-wide x tile).
// LDS: 16 rows x Kpad bf16 = im2col rows in K order -> A fragment is exactly
// two ds_load_b128 per 32-chunk. Waves pair output-channel tiles (ct, ct+4)
// so one A fragment feeds two WMMAs; the K loop is software-pipelined so the
// loads for chunk kc+1 are in flight while chunk kc's WMMAs execute.
// Conv bias omitted: training-mode BN cancels any per-channel constant exactly.
// ---------------------------------------------------------------------------
__global__ __launch_bounds__(128)
void rnet_conv3x3_wmma(const float* __restrict__ in, const unsigned short* __restrict__ wpk,
                       float* __restrict__ out, float* __restrict__ csum, float* __restrict__ csumsq,
                       int inC, int outC, int H, int W) {
  extern __shared__ unsigned short lds[];   // 16 * Kpad bf16
  const int x0   = blockIdx.x << 4;
  const int y    = blockIdx.y;
  const int b    = blockIdx.z;
  const int tid  = threadIdx.x;
  const int lane = tid & 31;
  const int wave = tid >> 5;
  const int K    = inC * 9;
  const int nKC  = (K + 31) >> 5;
  const int Kpad = nKC << 5;

  // ---- im2col staging: lds[m*Kpad + rp*inC + ic], coalesced over ic ----
  for (int rp = 0; rp < 9; ++rp) {
    const int ry = rp / 3, rx = rp % 3;       // constant-folded per iteration
    const int iy = y - 1 + ry;
    const bool rowok = (iy >= 0) && (iy < H);
    const int iyc = rowok ? iy : 0;
    const float* srow = in + ((size_t)(b * H + iyc) * W) * inC;
    for (int m = 0; m < 16; ++m) {
      const int ix = x0 + m - 1 + rx;
      const bool ok = rowok && (ix >= 0) && (ix < W);
      const int ixc = (ix >= 0) ? (ix < W ? ix : W - 1) : 0;  // clamp: no OOB addr
      const float* src = srow + (size_t)ixc * inC;
      unsigned short* dst = lds + m * Kpad + rp * inC;
      for (int ic = tid; ic < inC; ic += 128)
        dst[ic] = ok ? f32_to_bf16_rne(src[ic]) : (unsigned short)0;
    }
  }
  if (K < Kpad) {  // only enc1 (K=18 -> Kpad=32): zero the K padding tail
    for (int i = tid; i < 16 * 32; i += 128) {
      int m = i >> 5, k = i & 31;
      if (k >= K) lds[m * Kpad + k] = 0;
    }
  }
  __syncthreads();

  // ---- WMMA main loop ----
  const int m16   = lane & 15;             // A row = output pixel in tile / C col = channel
  const int khalf = (lane >> 4) << 3;      // 0 or 8: K-half (A) == M offset (C/D)
  const int nCT   = outC >> 4;
  const uint4* arow = (const uint4*)(lds + m16 * Kpad + khalf); // chunk kc at +kc*4, +kc*4+2

  union Frag { v16bf v; uint4 q[2]; };

  for (int ct0 = wave; ct0 < nCT; ct0 += 8) {
    const bool dual = (ct0 + 4) < nCT;     // wave-uniform: EXEC untouched
    v8f c0 = {0.f, 0.f, 0.f, 0.f, 0.f, 0.f, 0.f, 0.f};
    v8f c1 = {0.f, 0.f, 0.f, 0.f, 0.f, 0.f, 0.f, 0.f};
    const uint4* bq0 = (const uint4*)(wpk + (size_t)ct0 * nKC * 512 + lane * 16);
    const uint4* bq1 = dual ? (const uint4*)(wpk + (size_t)(ct0 + 4) * nKC * 512 + lane * 16)
                            : bq0;

    Frag A0, B0, B1, An, B0n, B1n;
    A0.q[0] = arow[0];  A0.q[1] = arow[2];
    B0.q[0] = bq0[0];   B0.q[1] = bq0[1];
    B1.q[0] = bq1[0];   B1.q[1] = bq1[1];

    for (int kc = 0; kc + 1 < nKC; ++kc) {
      // prefetch chunk kc+1 while chunk kc computes
      An.q[0]  = arow[(kc + 1) * 4];
      An.q[1]  = arow[(kc + 1) * 4 + 2];
      B0n.q[0] = bq0[(kc + 1) * 64];
      B0n.q[1] = bq0[(kc + 1) * 64 + 1];
      B1n.q[0] = bq1[(kc + 1) * 64];
      B1n.q[1] = bq1[(kc + 1) * 64 + 1];
      c0 = __builtin_amdgcn_wmma_f32_16x16x32_bf16(false, A0.v, false, B0.v,
                                                   (short)0, c0, false, false);
      if (dual)
        c1 = __builtin_amdgcn_wmma_f32_16x16x32_bf16(false, A0.v, false, B1.v,
                                                     (short)0, c1, false, false);
      A0 = An; B0 = B0n; B1 = B1n;
    }
    c0 = __builtin_amdgcn_wmma_f32_16x16x32_bf16(false, A0.v, false, B0.v,
                                                 (short)0, c0, false, false);
    if (dual)
      c1 = __builtin_amdgcn_wmma_f32_16x16x32_bf16(false, A0.v, false, B1.v,
                                                   (short)0, c1, false, false);

    // C/D layout: VGPR r -> pixel x0 + khalf + r, channel ct*16 + (lane&15).
    // NHWC store: 16 consecutive lanes = 16 consecutive channels (coalesced).
    {
      int n = ct0 * 16 + m16;
      float* op = out + ((size_t)((b * H + y) * W) + x0 + khalf) * outC + n;
      float s = 0.0f, s2 = 0.0f;
#pragma unroll
      for (int r = 0; r < 8; ++r) {
        float v = c0[r];
        op[(size_t)r * outC] = v;
        s += v; s2 += v * v;
      }
      atomicAdd(&csum[n],   s);
      atomicAdd(&csumsq[n], s2);
    }
    if (dual) {
      int n = (ct0 + 4) * 16 + m16;
      float* op = out + ((size_t)((b * H + y) * W) + x0 + khalf) * outC + n;
      float s = 0.0f, s2 = 0.0f;
#pragma unroll
      for (int r = 0; r < 8; ++r) {
        float v = c1[r];
        op[(size_t)r * outC] = v;
        s += v; s2 += v * v;
      }
      atomicAdd(&csum[n],   s);
      atomicAdd(&csumsq[n], s2);
    }
  }
}

// ---------------------------------------------------------------------------
// BN finalize + apply (+ReLU, in place, NHWC)
// ---------------------------------------------------------------------------
__global__ void rnet_bn_finalize(const float* __restrict__ sum, const float* __restrict__ sumsq,
                                 const float* __restrict__ g, const float* __restrict__ be,
                                 float* __restrict__ scale, float* __restrict__ shift,
                                 int C, float invN) {
  int c = blockIdx.x * blockDim.x + threadIdx.x;
  if (c >= C) return;
  float mean = sum[c] * invN;
  float var  = sumsq[c] * invN - mean * mean;
  float sc   = g[c] * rsqrtf(var + 1e-5f);
  scale[c] = sc;
  shift[c] = be[c] - mean * sc;
}

__global__ void rnet_bn_relu(float* __restrict__ x, const float* __restrict__ scale,
                             const float* __restrict__ shift, int Cmask, int total) {
  int i = blockIdx.x * blockDim.x + threadIdx.x;
  if (i >= total) return;
  int c = i & Cmask;                 // all outC are powers of two
  float v = x[i] * scale[c] + shift[c];
  x[i] = v > 0.0f ? v : 0.0f;
}

// ---------------------------------------------------------------------------
// 2x2 max pool (NHWC)
// ---------------------------------------------------------------------------
__global__ void rnet_maxpool2(const float* __restrict__ in, float* __restrict__ out,
                              int C, int H, int W) {
  int ow = W >> 1, oh = H >> 1;
  int i = blockIdx.x * blockDim.x + threadIdx.x;
  int total = 4 * C * oh * ow;
  if (i >= total) return;
  int c = i % C;
  int j = i / C;
  int x = j % ow;
  int y = (j / ow) % oh;
  int b = j / (ow * oh);
  const float* p = in + ((size_t)((b * H + 2 * y) * W) + 2 * x) * C + c;
  size_t rs = (size_t)W * C;
  out[i] = fmaxf(fmaxf(p[0], p[C]), fmaxf(p[rs], p[rs + C]));
}

// ---------------------------------------------------------------------------
// bilinear x2 upsample, align_corners=True (NHWC), writes concat channels [0,C)
// ---------------------------------------------------------------------------
__global__ void rnet_upsample2(const float* __restrict__ in, float* __restrict__ dst,
                               int C, int H, int W, int dstC) {
  int oh = 2 * H, ow = 2 * W;
  int i = blockIdx.x * blockDim.x + threadIdx.x;
  int total = 4 * C * oh * ow;
  if (i >= total) return;
  int c = i % C;
  int j = i / C;
  int x = j % ow;
  int y = (j / ow) % oh;
  int b = j / (ow * oh);
  float fy = (float)y * (float)(H - 1) / (float)(oh - 1);
  float fx = (float)x * (float)(W - 1) / (float)(ow - 1);
  int y0 = (int)floorf(fy); int y1 = y0 + 1 < H ? y0 + 1 : H - 1; float wy = fy - (float)y0;
  int xx0 = (int)floorf(fx); int xx1 = xx0 + 1 < W ? xx0 + 1 : W - 1; float wx = fx - (float)xx0;
  const float* p = in + ((size_t)b * H * W) * C + c;
  float v00 = p[((size_t)y0 * W + xx0) * C];
  float v01 = p[((size_t)y0 * W + xx1) * C];
  float v10 = p[((size_t)y1 * W + xx0) * C];
  float v11 = p[((size_t)y1 * W + xx1) * C];
  float t0 = v00 * (1.0f - wx) + v01 * wx;
  float t1 = v10 * (1.0f - wx) + v11 * wx;
  dst[((size_t)(b * oh + y) * ow + x) * dstC + c] = t0 * (1.0f - wy) + t1 * wy;
}

// skip-connection copy into concat buffer at channel offset (NHWC)
__global__ void rnet_copy_ch(const float* __restrict__ src, float* __restrict__ dst,
                             int C, int dstC, int off, int total) {
  int i = blockIdx.x * blockDim.x + threadIdx.x;
  if (i >= total) return;
  int c = i % C;
  int j = i / C;                       // b*HW + pix (same flattening in dst)
  dst[(size_t)j * dstC + off + c] = src[i];
}

// ---------------------------------------------------------------------------
// 1x1 heads (NHWC d1, 32ch): det=sigmoid, vel=tanh(x2), snr=relu.
// Outputs written in the reference's NCHW layout (det[b,hw], vel[b,2,hw], snr[b,hw]).
// ---------------------------------------------------------------------------
__global__ __launch_bounds__(256)
void rnet_heads(const float* __restrict__ d1,
                const float* __restrict__ wd, const float* __restrict__ bd,
                const float* __restrict__ wv, const float* __restrict__ bv,
                const float* __restrict__ wsn, const float* __restrict__ bsn,
                float* __restrict__ det, float* __restrict__ vel, float* __restrict__ snr,
                int HW) {
  __shared__ float w[128];
  __shared__ float bias[4];
  if (threadIdx.x < 32) {
    w[threadIdx.x]      = wd[threadIdx.x];
    w[32 + threadIdx.x] = wv[threadIdx.x];
    w[64 + threadIdx.x] = wv[32 + threadIdx.x];
    w[96 + threadIdx.x] = wsn[threadIdx.x];
  }
  if (threadIdx.x == 0) { bias[0] = bd[0]; bias[1] = bv[0]; bias[2] = bv[1]; bias[3] = bsn[0]; }
  __syncthreads();
  int i = blockIdx.x * blockDim.x + threadIdx.x;   // pixel id, (b,y,x) flattened
  if (i >= 4 * HW) return;
  const float4* p4 = (const float4*)(d1 + (size_t)i * 32);
  float a0 = bias[0], a1 = bias[1], a2 = bias[2], a3 = bias[3];
#pragma unroll
  for (int q = 0; q < 8; ++q) {
    float4 v = p4[q];
    int c = q * 4;
    a0 += v.x * w[c]      + v.y * w[c + 1]      + v.z * w[c + 2]      + v.w * w[c + 3];
    a1 += v.x * w[32 + c] + v.y * w[33 + c]     + v.z * w[34 + c]     + v.w * w[35 + c];
    a2 += v.x * w[64 + c] + v.y * w[65 + c]     + v.z * w[66 + c]     + v.w * w[67 + c];
    a3 += v.x * w[96 + c] + v.y * w[97 + c]     + v.z * w[98 + c]     + v.w * w[99 + c];
  }
  int hw = i % HW;
  int b  = i / HW;
  det[i] = 1.0f / (1.0f + expf(-a0));
  vel[(size_t)b * 2 * HW + hw]      = tanhf(a1);
  vel[(size_t)b * 2 * HW + HW + hw] = tanhf(a2);
  snr[i] = fmaxf(a3, 0.0f);
}

// ---------------------------------------------------------------------------
// Per-batch top-10 decode. One block per batch (256 threads).
// ---------------------------------------------------------------------------
__global__ __launch_bounds__(256)
void rnet_topk_decode(const float* __restrict__ det, const float* __restrict__ vel,
                      const float* __restrict__ snr, float* __restrict__ tail,
                      int R, int N) {
  __shared__ float sp[256 * TOPK];
  __shared__ int   si[256 * TOPK];
  __shared__ float rv[256];
  __shared__ int   ri[256];
  __shared__ float selP[TOPK];
  __shared__ int   selI[TOPK];
  int b = blockIdx.x;
  int t = threadIdx.x;
  float lp[TOPK]; int li[TOPK];
#pragma unroll
  for (int j = 0; j < TOPK; ++j) { lp[j] = -1e30f; li[j] = 0; }
  const float* dmap = det + (size_t)b * N;
  for (int i = t; i < N; i += 256) {
    float v = dmap[i];
    if (v > lp[TOPK - 1]) {
      int j = TOPK - 1;
      while (j > 0 && lp[j - 1] < v) { lp[j] = lp[j - 1]; li[j] = li[j - 1]; --j; }
      lp[j] = v; li[j] = i;
    }
  }
#pragma unroll
  for (int j = 0; j < TOPK; ++j) { sp[t * TOPK + j] = lp[j]; si[t * TOPK + j] = li[j]; }
  __syncthreads();

  for (int round = 0; round < TOPK; ++round) {
    float bv = -1e30f; int bi = t * TOPK;
#pragma unroll
    for (int j = 0; j < TOPK; ++j) {
      float v = sp[t * TOPK + j];
      if (v > bv) { bv = v; bi = t * TOPK + j; }
    }
    rv[t] = bv; ri[t] = bi;
    __syncthreads();
    for (int s = 128; s > 0; s >>= 1) {
      if (t < s) { if (rv[t + s] > rv[t]) { rv[t] = rv[t + s]; ri[t] = ri[t + s]; } }
      __syncthreads();
    }
    if (t == 0) { selP[round] = rv[0]; selI[round] = si[ri[0]]; sp[ri[0]] = -1e30f; }
    __syncthreads();
  }

  if (t < TOPK) {
    float p = selP[t];
    int idx = selI[t];
    int d = idx / R, r = idx % R;
    const float* velb = vel + (size_t)b * 2 * N;
    float vx = velb[idx];
    float vy = velb[(size_t)N + idx];
    float sv = snr[(size_t)b * N + idx];
    float sdb = 10.0f * log10f(sv + 1e-10f);
    int o = b * TOPK + t;
    tail[0 * 40 + o] = (float)d;
    tail[1 * 40 + o] = (float)r;
    tail[2 * 40 + o] = p;
    tail[3 * 40 + o] = vx;
    tail[4 * 40 + o] = vy;
    tail[5 * 40 + o] = sdb;
    tail[6 * 40 + o] = (p > 0.5f) ? 1.f : 0.f;
  }
}

// ---------------------------------------------------------------------------
// Host-side launch
// ---------------------------------------------------------------------------
extern "C" void kernel_launch(void* const* d_in, const int* in_sizes, int n_in,
                              void* d_out, int out_size, void* d_ws, size_t ws_size,
                              hipStream_t stream) {
  (void)in_sizes; (void)n_in; (void)out_size; (void)ws_size;
  const int B = 4;

  // ---- input pointers (setup_inputs insertion order; blocks: w,b,g,be) ----
  const float* x = (const float*)d_in[0];
  const float* enc1_w = (const float*)d_in[1];  const float* enc1_g = (const float*)d_in[3];  const float* enc1_be = (const float*)d_in[4];
  const float* enc2_w = (const float*)d_in[5];  const float* enc2_g = (const float*)d_in[7];  const float* enc2_be = (const float*)d_in[8];
  const float* enc3_w = (const float*)d_in[9];  const float* enc3_g = (const float*)d_in[11]; const float* enc3_be = (const float*)d_in[12];
  const float* enc4_w = (const float*)d_in[13]; const float* enc4_g = (const float*)d_in[15]; const float* enc4_be = (const float*)d_in[16];
  const float* dec3_w = (const float*)d_in[17]; const float* dec3_g = (const float*)d_in[19]; const float* dec3_be = (const float*)d_in[20];
  const float* dec2_w = (const float*)d_in[21]; const float* dec2_g = (const float*)d_in[23]; const float* dec2_be = (const float*)d_in[24];
  const float* dec1_w = (const float*)d_in[25]; const float* dec1_g = (const float*)d_in[27]; const float* dec1_be = (const float*)d_in[28];
  const float* det_w  = (const float*)d_in[29]; const float* det_b  = (const float*)d_in[30];
  const float* vel_w  = (const float*)d_in[31]; const float* vel_b  = (const float*)d_in[32];
  const float* snr_w  = (const float*)d_in[33]; const float* snr_b  = (const float*)d_in[34];

  // ---- workspace bump allocator (256B aligned) ----
  char* base = (char*)d_ws;
  size_t off = 0;
  auto alloc = [&](size_t bytes) -> void* {
    void* p = base + off;
    off += (bytes + 255) & ~(size_t)255;
    return p;
  };
  auto allocf = [&](size_t n) -> float* { return (float*)alloc(n * sizeof(float)); };
  auto alloch = [&](size_t n) -> unsigned short* { return (unsigned short*)alloc(n * sizeof(unsigned short)); };

  // all activations NHWC
  float* xh   = allocf((size_t)B * 2 * 512 * 512);
  float* e1   = allocf((size_t)B * 32 * 512 * 512);
  float* p1   = allocf((size_t)B * 32 * 256 * 256);
  float* e2   = allocf((size_t)B * 64 * 256 * 256);
  float* p2   = allocf((size_t)B * 64 * 128 * 128);
  float* e3   = allocf((size_t)B * 128 * 128 * 128);
  float* p3   = allocf((size_t)B * 128 * 64 * 64);
  float* e4   = allocf((size_t)B * 256 * 64 * 64);
  float* cat3 = allocf((size_t)B * 384 * 128 * 128);
  float* d3   = allocf((size_t)B * 128 * 128 * 128);
  float* cat2 = allocf((size_t)B * 192 * 256 * 256);
  float* d2   = allocf((size_t)B * 64 * 256 * 256);
  float* cat1 = allocf((size_t)B * 96 * 512 * 512);
  float* d1   = allocf((size_t)B * 32 * 512 * 512);
  float* csum   = allocf(256);
  float* csumsq = allocf(256);
  float* scale  = allocf(256);
  float* shift  = allocf(256);

  auto kpad = [](int inC) { return ((inC * 9 + 31) / 32) * 32; };
  unsigned short* wpk1  = alloch((size_t)32  * kpad(2));
  unsigned short* wpk2  = alloch((size_t)64  * kpad(32));
  unsigned short* wpk3  = alloch((size_t)128 * kpad(64));
  unsigned short* wpk4  = alloch((size_t)256 * kpad(128));
  unsigned short* wpkD3 = alloch((size_t)128 * kpad(384));
  unsigned short* wpkD2 = alloch((size_t)64  * kpad(192));
  unsigned short* wpkD1 = alloch((size_t)32  * kpad(96));

  // ---- conv block driver ----
  auto run_block = [&](const float* in, int inC, int H, int W, int outC,
                       const float* w, const float* g, const float* be,
                       unsigned short* wpk, float* out) {
    int K = inC * 9;
    int nKC = (K + 31) >> 5;
    int ptot = (outC / 16) * nKC * 512;
    rnet_pack_w<<<(ptot + 255) / 256, 256, 0, stream>>>(w, wpk, inC, nKC, outC);
    rnet_zero_f32<<<1, 256, 0, stream>>>(csum, 256);
    rnet_zero_f32<<<1, 256, 0, stream>>>(csumsq, 256);
    dim3 grid(W / 16, H, B);
    size_t shmem = (size_t)16 * (nKC << 5) * sizeof(unsigned short);
    rnet_conv3x3_wmma<<<grid, 128, shmem, stream>>>(in, wpk, out, csum, csumsq,
                                                    inC, outC, H, W);
    rnet_bn_finalize<<<(outC + 63) / 64, 64, 0, stream>>>(csum, csumsq, g, be, scale, shift,
                                                          outC, 1.0f / (float)(B * H * W));
    int tot = B * outC * H * W;
    rnet_bn_relu<<<(tot + 255) / 256, 256, 0, stream>>>(out, scale, shift, outC - 1, tot);
  };
  auto run_pool = [&](const float* in, float* out, int C, int H, int W) {
    int tot = B * C * (H / 2) * (W / 2);
    rnet_maxpool2<<<(tot + 255) / 256, 256, 0, stream>>>(in, out, C, H, W);
  };
  auto run_up = [&](const float* in, float* dst, int C, int H, int W, int dstC) {
    int tot = B * C * 4 * H * W;
    rnet_upsample2<<<(tot + 255) / 256, 256, 0, stream>>>(in, dst, C, H, W, dstC);
  };
  auto run_copy = [&](const float* src, float* dst, int C, int HW, int dstC, int choff) {
    int tot = B * C * HW;
    rnet_copy_ch<<<(tot + 255) / 256, 256, 0, stream>>>(src, dst, C, dstC, choff, tot);
  };

  // ---- input to NHWC ----
  {
    int tot = B * 2 * 512 * 512;
    rnet_nchw_to_nhwc<<<(tot + 255) / 256, 256, 0, stream>>>(x, xh, 2, 512 * 512);
  }

  // ---- encoder ----
  run_block(xh, 2,   512, 512, 32,  enc1_w, enc1_g, enc1_be, wpk1, e1);
  run_pool(e1, p1, 32, 512, 512);
  run_block(p1, 32,  256, 256, 64,  enc2_w, enc2_g, enc2_be, wpk2, e2);
  run_pool(e2, p2, 64, 256, 256);
  run_block(p2, 64,  128, 128, 128, enc3_w, enc3_g, enc3_be, wpk3, e3);
  run_pool(e3, p3, 128, 128, 128);
  run_block(p3, 128, 64,  64,  256, enc4_w, enc4_g, enc4_be, wpk4, e4);

  // ---- decoder ----
  run_up(e4, cat3, 256, 64, 64, 384);
  run_copy(e3, cat3, 128, 128 * 128, 384, 256);
  run_block(cat3, 384, 128, 128, 128, dec3_w, dec3_g, dec3_be, wpkD3, d3);

  run_up(d3, cat2, 128, 128, 128, 192);
  run_copy(e2, cat2, 64, 256 * 256, 192, 128);
  run_block(cat2, 192, 256, 256, 64, dec2_w, dec2_g, dec2_be, wpkD2, d2);

  run_up(d2, cat1, 64, 256, 256, 96);
  run_copy(e1, cat1, 32, 512 * 512, 96, 64);
  run_block(cat1, 96, 512, 512, 32, dec1_w, dec1_g, dec1_be, wpkD1, d1);

  // ---- heads write directly into d_out ----
  const int HW = 512 * 512;
  float* out_det  = (float*)d_out;                       // B*HW
  float* out_vel  = out_det + (size_t)B * HW;            // B*2*HW
  float* out_snr  = out_vel + (size_t)B * 2 * HW;        // B*HW
  float* out_tail = out_snr + (size_t)B * HW;            // 7 * 40 floats
  {
    int tot = B * HW;
    rnet_heads<<<(tot + 255) / 256, 256, 0, stream>>>(
        d1, det_w, det_b, vel_w, vel_b, snr_w, snr_b,
        out_det, out_vel, out_snr, HW);
  }

  // ---- per-batch top-10 decode ----
  rnet_topk_decode<<<B, 256, 0, stream>>>(out_det, out_vel, out_snr, out_tail, 512, HW);
}
